// SDPA_36112085024790
// MI455X (gfx1250) — matrix operand order repaired and checked by reading, not compiled, not run on hardware
//
#include <hip/hip_runtime.h>

// Problem constants (match reference)
#define Bc 4
#define Cc 128
#define Nn 4096   // H*W = 64*64

typedef __attribute__((ext_vector_type(16))) _Float16 v16h;
typedef __attribute__((ext_vector_type(8)))  float    v8f;
typedef __attribute__((ext_vector_type(4)))  float    f32x4;

union Frag {
  f32x4    q[2];
  v16h     h;
  _Float16 e[16];
};

__device__ __forceinline__ v8f wmma16(v16h a, v16h b, v8f c) {
  // D(16x16,f32) = A(16x32,f16) x B(32x16,f16) + C
  return __builtin_amdgcn_wmma_f32_16x16x32_f16(
      /*neg_a=*/false, a, /*neg_b=*/false, b,
      /*c_mod=*/(short)0, c, /*reuse_a=*/false, /*reuse_b=*/false);
}

// -------------------------------------------------------------------------
// Kernel 1: QKV projection via WMMA.
//   q[b,n,d] = sum_c x[b,c,n] * Wq[d,c] + bq[d]   (ditto k, v)
// Computes Out^T = W * X per tile (D rows = out-channel, cols = token).
// Writes Qh (pre-scaled by 1/sqrt(C)) and Kh token-major [B][N][C] f16,
// and Vt channel-major [B][C][N] f16.
// -------------------------------------------------------------------------
__global__ __launch_bounds__(256) void qkv_proj(
    const float* __restrict__ x,
    const float* __restrict__ Wq, const float* __restrict__ bq,
    const float* __restrict__ Wk, const float* __restrict__ bk,
    const float* __restrict__ Wv, const float* __restrict__ bv,
    _Float16* __restrict__ Qh, _Float16* __restrict__ Kh,
    _Float16* __restrict__ Vt)
{
  const int b    = blockIdx.y;
  const int n0   = blockIdx.x * 64;          // 64 tokens per block
  const int tid  = threadIdx.x;
  const int lane = tid & 31;
  const int l15  = lane & 15;
  const int lh   = (lane >> 4) & 1;
  const int wv   = tid >> 5;                 // 8 waves

  constexpr int PADX = 136;                  // f16 row stride (272B, 16B aligned)
  __shared__ _Float16 xs[64 * PADX];         // x tile transposed: [token][channel]

  // ---- stage + transpose x tile (f32 -> f16) ----
  {
    const int c  = tid >> 1;                 // channel 0..127
    const int tp = (tid & 1) * 32;           // token half
    const float* src = x + (size_t)(b * Cc + c) * Nn + n0 + tp;
    #pragma unroll
    for (int j4 = 0; j4 < 8; ++j4) {
      f32x4 v = *(const f32x4*)(src + j4 * 4);
      #pragma unroll
      for (int k = 0; k < 4; ++k)
        xs[(tp + j4 * 4 + k) * PADX + c] = (_Float16)v[k];
    }
  }
  __syncthreads();

  const float scale = 0.08838834764831845f;  // C^-0.5, folded into Q

  // 96 output tiles per block: 3 mats x 8 d-tiles x 4 token-tiles
  for (int tile = wv; tile < 96; tile += 8) {
    const int mat = tile % 3;
    const int r2  = tile / 3;
    const int d0  = (r2 & 7) * 16;
    const int t0  = (r2 >> 3) * 16;
    const float* Wm = (mat == 0) ? Wq : (mat == 1) ? Wk : Wv;
    const float* bm = (mat == 0) ? bq : (mat == 1) ? bk : bv;

    v8f acc;
    #pragma unroll
    for (int r = 0; r < 8; ++r) acc[r] = bm[d0 + r + lh * 8];

    const int drow = d0 + l15;
    #pragma unroll
    for (int cc = 0; cc < 4; ++cc) {
      const int c0 = cc * 32;
      // A: W tile 16(d) x 32(c); elems 0..7 = c0+lh*8.., elems 8..15 = +16
      Frag a;
      const float* wp = Wm + drow * Cc + c0 + lh * 8;
      f32x4 w0 = *(const f32x4*)(wp);
      f32x4 w1 = *(const f32x4*)(wp + 4);
      f32x4 w2 = *(const f32x4*)(wp + 16);
      f32x4 w3 = *(const f32x4*)(wp + 20);
      #pragma unroll
      for (int i = 0; i < 4; ++i) {
        a.e[i]      = (_Float16)w0[i];
        a.e[4 + i]  = (_Float16)w1[i];
        a.e[8 + i]  = (_Float16)w2[i];
        a.e[12 + i] = (_Float16)w3[i];
      }
      // B: X tile 32(c) x 16(token); elems 0..7 = c0+lh*8.., elems 8..15 = +16
      Frag bx;
      const _Float16* xp = &xs[(t0 + l15) * PADX + c0 + lh * 8];
      bx.q[0] = *(const f32x4*)(xp);
      bx.q[1] = *(const f32x4*)(xp + 16);
      acc = wmma16(a.h, bx.h, acc);
    }

    const int token = n0 + t0 + l15;
    if (mat == 2) {
      // V channel-major
      #pragma unroll
      for (int r = 0; r < 8; ++r)
        Vt[(size_t)(b * Cc + d0 + r + lh * 8) * Nn + token] = (_Float16)acc[r];
    } else {
      _Float16* outp = mat ? Kh : Qh;
      const float s = mat ? 1.0f : scale;
      #pragma unroll
      for (int r = 0; r < 8; ++r)
        outp[(size_t)(b * Nn + token) * Cc + d0 + r + lh * 8] =
            (_Float16)(acc[r] * s);
    }
  }
}

// -------------------------------------------------------------------------
// Kernel 2: flash attention. Each wave owns 16 queries; computes
// S^T = K * Q^T (queries on lanes -> cheap online softmax) and
// O^T += V^T * P (O^T is channel-major == required output layout).
// 4 waves share LDS-staged 32-key K / V^T tiles.
// -------------------------------------------------------------------------
__global__ __launch_bounds__(128) void attn(
    const _Float16* __restrict__ Qh, const _Float16* __restrict__ Kh,
    const _Float16* __restrict__ Vt, float* __restrict__ out)
{
  const int b    = blockIdx.y;
  const int tid  = threadIdx.x;
  const int lane = tid & 31;
  const int l15  = lane & 15;
  const int lh   = (lane >> 4) & 1;
  const int wv   = tid >> 5;
  const int q0   = blockIdx.x * 64 + wv * 16;   // wave's query base

  constexpr int PADK = 136;   // f16 row stride for K tile rows (272B)
  constexpr int PADV = 40;    // f16 row stride for V^T tile rows (80B)
  __shared__ _Float16 Klds[32 * PADK];    // [key][channel]
  __shared__ _Float16 Vlds[128 * PADV];   // [channel][key]

  // Persistent Q^T B-fragments (scale already folded in)
  Frag qf[4];
  {
    const _Float16* qp = Qh + (size_t)(b * Nn + q0 + l15) * Cc + lh * 8;
    #pragma unroll
    for (int cc = 0; cc < 4; ++cc) {
      qf[cc].q[0] = *(const f32x4*)(qp + cc * 32);
      qf[cc].q[1] = *(const f32x4*)(qp + cc * 32 + 16);
    }
  }

  const v8f vzero = {0.f, 0.f, 0.f, 0.f, 0.f, 0.f, 0.f, 0.f};
  v8f acc[8];
  #pragma unroll
  for (int i = 0; i < 8; ++i) acc[i] = vzero;

  float m_run = -3.0e38f;
  float l_run = 0.0f;

  for (int kb = 0; kb < Nn; kb += 32) {
    __syncthreads();   // previous compute done before LDS overwrite
    // ---- stage K tile [32 keys][128 ch] ----
    {
      int idx = tid;
      #pragma unroll
      for (int it = 0; it < 4; ++it, idx += 128) {
        const int row = idx >> 4, seg = idx & 15;
        f32x4 v = *(const f32x4*)(Kh + (size_t)(b * Nn + kb + row) * Cc + seg * 8);
        *(f32x4*)(&Klds[row * PADK + seg * 8]) = v;
      }
      // ---- stage V^T tile [128 ch][32 keys] ----
      idx = tid;
      #pragma unroll
      for (int it = 0; it < 4; ++it, idx += 128) {
        const int ch = idx >> 2, seg = idx & 3;
        f32x4 v = *(const f32x4*)(Vt + (size_t)(b * Cc + ch) * Nn + kb + seg * 8);
        *(f32x4*)(&Vlds[ch * PADV + seg * 8]) = v;
      }
    }
    __syncthreads();

    // ---- S^T tiles: keys kb..kb+15 (st0) and kb+16..kb+31 (st1) ----
    v8f st0 = vzero, st1 = vzero;
    #pragma unroll
    for (int cc = 0; cc < 4; ++cc) {
      const int c0 = cc * 32;
      Frag a0, a1;
      const _Float16* kp0 = &Klds[l15 * PADK + c0 + lh * 8];
      a0.q[0] = *(const f32x4*)(kp0);
      a0.q[1] = *(const f32x4*)(kp0 + 16);
      const _Float16* kp1 = &Klds[(16 + l15) * PADK + c0 + lh * 8];
      a1.q[0] = *(const f32x4*)(kp1);
      a1.q[1] = *(const f32x4*)(kp1 + 16);
      st0 = wmma16(a0.h, qf[cc].h, st0);
      st1 = wmma16(a1.h, qf[cc].h, st1);
    }

    // ---- online softmax (queries on lanes; one xor-16 shuffle) ----
    float mloc = st0[0];
    #pragma unroll
    for (int i = 1; i < 8; ++i) mloc = fmaxf(mloc, st0[i]);
    #pragma unroll
    for (int i = 0; i < 8; ++i) mloc = fmaxf(mloc, st1[i]);
    const float mc    = fmaxf(mloc, __shfl_xor(mloc, 16, 32));
    const float m_new = fmaxf(m_run, mc);
    const float alpha = __expf(m_run - m_new);

    float e0[8], e1[8], s_loc = 0.0f;
    #pragma unroll
    for (int i = 0; i < 8; ++i) { e0[i] = __expf(st0[i] - m_new); s_loc += e0[i]; }
    #pragma unroll
    for (int i = 0; i < 8; ++i) { e1[i] = __expf(st1[i] - m_new); s_loc += e1[i]; }
    l_run = l_run * alpha + s_loc + __shfl_xor(s_loc, 16, 32);
    m_run = m_new;

    #pragma unroll
    for (int i = 0; i < 8; ++i)
      #pragma unroll
      for (int r = 0; r < 8; ++r) acc[i][r] *= alpha;

    // ---- relayout P^T (D-layout) -> P B-fragment (32 keys x 16 queries) ----
    // B layout: lanes 0-15 hold keys 0..15, lanes 16-31 hold keys 16..31.
    float x0[8], x1[8];
    #pragma unroll
    for (int i = 0; i < 8; ++i) {
      x0[i] = __shfl_xor(e0[i], 16, 32);   // lo lanes receive keys 8..15
      x1[i] = __shfl_xor(e1[i], 16, 32);   // hi lanes receive keys 16..23
    }
    Frag pf;
    #pragma unroll
    for (int j = 0; j < 8; ++j) {
      pf.e[j]     = (_Float16)(lh ? x1[j] : e0[j]);   // K = lh*16 + j
      pf.e[8 + j] = (_Float16)(lh ? e1[j] : x0[j]);   // K = lh*16 + 8 + j
    }

    // ---- O^T += V^T x P ----
    #pragma unroll
    for (int ct = 0; ct < 8; ++ct) {
      Frag va;
      const _Float16* vp = &Vlds[(ct * 16 + l15) * PADV + lh * 16];
      va.q[0] = *(const f32x4*)(vp);
      va.q[1] = *(const f32x4*)(vp + 8);
      acc[ct] = wmma16(va.h, pf.h, acc[ct]);
    }
  }

  // ---- epilogue: normalize, store to [B][C][H*W] fp32 (channel-major) ----
  const float rinv = 1.0f / l_run;
  #pragma unroll
  for (int ct = 0; ct < 8; ++ct) {
    #pragma unroll
    for (int r = 0; r < 8; ++r) {
      const int ch = ct * 16 + r + lh * 8;
      out[(size_t)(b * Cc + ch) * Nn + q0 + l15] = acc[ct][r] * rinv;
    }
  }
}

// -------------------------------------------------------------------------
extern "C" void kernel_launch(void* const* d_in, const int* in_sizes, int n_in,
                              void* d_out, int out_size, void* d_ws, size_t ws_size,
                              hipStream_t stream) {
  (void)in_sizes; (void)n_in; (void)out_size; (void)ws_size;
  const float* x  = (const float*)d_in[0];
  const float* Wq = (const float*)d_in[1];
  const float* bq = (const float*)d_in[2];
  const float* Wk = (const float*)d_in[3];
  const float* bk = (const float*)d_in[4];
  const float* Wv = (const float*)d_in[5];
  const float* bv = (const float*)d_in[6];
  float* out = (float*)d_out;

  // Workspace: Qh, Kh token-major; Vt channel-major (f16) -> 12 MB total
  _Float16* Qh = (_Float16*)d_ws;
  _Float16* Kh = Qh + (size_t)Bc * Nn * Cc;
  _Float16* Vt = Kh + (size_t)Bc * Nn * Cc;

  dim3 g1(Nn / 64, Bc);
  qkv_proj<<<g1, 256, 0, stream>>>(x, Wq, bq, Wk, bk, Wv, bv, Qh, Kh, Vt);

  dim3 g2(Nn / 64, Bc);
  attn<<<g2, 128, 0, stream>>>(Qh, Kh, Vt, out);
}
